// SemiTemplateGNN_66279935312490
// MI455X (gfx1250) — matrix operand order repaired and checked by reading, not compiled, not run on hardware
//
#include <hip/hip_runtime.h>

// ---------------------------------------------------------------------------
// SemiTemplateGNN (3-layer GAT) for gfx1250 / MI455X.
// Dense projections use V_WMMA_F32_16X16X32_BF16 (bf16 in, f32 accumulate).
// Edge softmax/aggregation uses L2-resident gathers + float atomics
// (entire per-layer state < 192 MB L2).
// ---------------------------------------------------------------------------

typedef __attribute__((ext_vector_type(16))) __bf16      v16bf;
typedef __attribute__((ext_vector_type(8)))  float       v8f;
typedef __attribute__((ext_vector_type(4)))  unsigned int u32x4;

union FragBF { v16bf v; u32x4 q[2]; };

__device__ __forceinline__ unsigned short f2bf(float f) {
  unsigned int u = __float_as_uint(f);
  u += 0x7FFFu + ((u >> 16) & 1u);          // round-to-nearest-even
  return (unsigned short)(u >> 16);
}

// ----------------------------- utility ------------------------------------
__global__ void k_fill(float* __restrict__ p, float v, int n) {
  int i = blockIdx.x * blockDim.x + threadIdx.x;
  if (i < n) p[i] = v;
}

// feat0 = concat(x, emb[class_label[batch[n]]])  -> bf16 [N,128]
__global__ void k_concat(const float* __restrict__ x, const int* __restrict__ batch,
                         const int* __restrict__ clab, const float* __restrict__ emb,
                         unsigned short* __restrict__ featB, int N) {
  int i = blockIdx.x * blockDim.x + threadIdx.x;
  if (i >= N * 128) return;
  int n = i >> 7, c = i & 127;
  float v = (c < 64) ? x[n * 64 + c]
                     : emb[clab[batch[n]] * 64 + (c - 64)];
  featB[i] = f2bf(v);
}

__global__ void k_cvt_bf16(const float* __restrict__ src,
                           unsigned short* __restrict__ dst, int n) {
  int i = blockIdx.x * blockDim.x + threadIdx.x;
  if (i < n) dst[i] = f2bf(src[i]);
}

// ----------------------------- WMMA GEMM ----------------------------------
// C[M,NOUT] = A[M,K](bf16) * B[NOUT,K](bf16, i.e. W row-major == B^T)
// One wave computes one 16x16 C tile; K stepped by 32 per v_wmma issue.
__global__ void k_gemm_bf16(const unsigned short* __restrict__ A,
                            const unsigned short* __restrict__ B,
                            float* __restrict__ C,
                            int M, int K, int NOUT) {
  const int lane   = threadIdx.x & 31;
  const int wave   = (blockIdx.x * blockDim.x + threadIdx.x) >> 5;
  const int ntiles = NOUT >> 4;
  const int mtile  = wave / ntiles;
  const int ntile  = wave - mtile * ntiles;
  if (mtile * 16 >= M) return;                    // wave-uniform: EXEC stays all-1

  const int l15 = lane & 15;
  const int hi  = lane >> 4;                      // half-wave select
  // A: lane holds row (mtile*16 + l15), K-slices {hi*8 .. hi*8+7, 16+hi*8 ..}
  const unsigned short* aBase = A + (size_t)(mtile * 16 + l15) * K + (hi << 3);
  // B: lane holds column (ntile*16 + l15) of W^T, K-slices {hi*8 .. hi*8+7, 16+hi*8 ..}
  const unsigned short* bBase = B + (size_t)(ntile * 16 + l15) * K + (hi << 3);

  v8f acc = {};
  for (int kb = 0; kb < K; kb += 32) {
    FragBF fa, fb;
    fa.q[0] = *(const u32x4*)(aBase + kb);        // K = kb + hi*8 + 0..7
    fa.q[1] = *(const u32x4*)(aBase + kb + 16);   // K = kb + 16 + hi*8 + 0..7
    fb.q[0] = *(const u32x4*)(bBase + kb);        // K = kb + hi*8 + 0..7
    fb.q[1] = *(const u32x4*)(bBase + kb + 16);   // K = kb + 16 + hi*8 + 0..7
    acc = __builtin_amdgcn_wmma_f32_16x16x32_bf16(
        /*neg_a=*/false, fa.v, /*neg_b=*/false, fb.v,
        /*c_mod=*/(short)0, acc, /*reuse_a=*/false, /*reuse_b=*/false);
  }
  // C/D layout: VGPR v, lane L -> (M = v + 8*(L>=16), N = L&15)
  float* cBase = C + (size_t)(mtile * 16 + (hi << 3)) * NOUT + ntile * 16 + l15;
#pragma unroll
  for (int r = 0; r < 8; ++r) cBase[(size_t)r * NOUT] = acc[r];
}

// ----------------------------- attention scores ---------------------------
// s_src[n,h] = <h[n,h,:], a_src[h,:]>, same for dst. One wave per (node,head).
__global__ void k_scores(const float* __restrict__ h, const float* __restrict__ asrc,
                         const float* __restrict__ adst, float* __restrict__ ssrc,
                         float* __restrict__ sdst, int N) {
  int w    = (blockIdx.x * blockDim.x + threadIdx.x) >> 5;
  int lane = threadIdx.x & 31;
  if (w >= N * 4) return;
  int n = w >> 2, hh = w & 3;
  const float* hv = h + (size_t)n * 256 + hh * 64;
  const float* as = asrc + hh * 64;
  const float* ad = adst + hh * 64;
  float v0 = hv[lane], v1 = hv[lane + 32];
  float ps = v0 * as[lane] + v1 * as[lane + 32];
  float pd = v0 * ad[lane] + v1 * ad[lane + 32];
  for (int o = 16; o > 0; o >>= 1) {
    ps += __shfl_down(ps, o);
    pd += __shfl_down(pd, o);
  }
  if (lane == 0) { ssrc[w] = ps; sdst[w] = pd; }
}

// ----------------------------- edge helpers -------------------------------
__device__ __forceinline__ void edge_ends(const int* __restrict__ ei, int e, int E,
                                          int& s, int& d) {
  if (e < E) { s = ei[e]; d = ei[E + e]; }
  else       { s = e - E; d = e - E; }            // self loops
}

__device__ __forceinline__ float lrelu02(float a) { return a > 0.f ? a : 0.2f * a; }

__device__ __forceinline__ void atomic_maxf(float* addr, float val) {
  int* ai  = (int*)addr;
  int  cur = __float_as_int(*addr);
  while (__int_as_float(cur) < val) {
    int prev = atomicCAS(ai, cur, __float_as_int(val));
    if (prev == cur) break;
    cur = prev;
  }
}

__global__ void k_alpha_max(const int* __restrict__ ei, const float* __restrict__ ssrc,
                            const float* __restrict__ sdst, float* __restrict__ m,
                            int E, int total) {
  int i = blockIdx.x * blockDim.x + threadIdx.x;
  if (i >= total) return;
  int e = i >> 2, hh = i & 3;
  int s, d; edge_ends(ei, e, E, s, d);
  float al = lrelu02(ssrc[s * 4 + hh] + sdst[d * 4 + hh]);
  atomic_maxf(&m[d * 4 + hh], al);
}

__global__ void k_exp_sum(const int* __restrict__ ei, const float* __restrict__ ssrc,
                          const float* __restrict__ sdst, const float* __restrict__ m,
                          float* __restrict__ denom, int E, int total) {
  int i = blockIdx.x * blockDim.x + threadIdx.x;
  if (i >= total) return;
  int e = i >> 2, hh = i & 3;
  int s, d; edge_ends(ei, e, E, s, d);
  float al = lrelu02(ssrc[s * 4 + hh] + sdst[d * 4 + hh]);
  float mm = m[d * 4 + hh];
  if (mm < -1e37f) mm = 0.f;                      // isolated-node safety
  atomicAdd(&denom[d * 4 + hh], __expf(al - mm));
}

// out[dst] += h[src] * softmax_weight. One 256-thread block per edge;
// gathers hit the 192MB L2 (h fits entirely).
__global__ void k_aggregate(const int* __restrict__ ei, const float* __restrict__ ssrc,
                            const float* __restrict__ sdst, const float* __restrict__ m,
                            const float* __restrict__ denom, const float* __restrict__ h,
                            float* __restrict__ out, int E) {
  int e  = blockIdx.x;
  int c  = threadIdx.x;                           // 0..255
  int hh = c >> 6;
  int s, d; edge_ends(ei, e, E, s, d);
  float al = lrelu02(ssrc[s * 4 + hh] + sdst[d * 4 + hh]);
  float mm = m[d * 4 + hh];
  if (mm < -1e37f) mm = 0.f;
  float w = __expf(al - mm) / (denom[d * 4 + hh] + 1e-16f);
  atomicAdd(&out[(size_t)d * 256 + c], h[(size_t)s * 256 + c] * w);
}

// feat = leaky_relu(feat + b, 0.01) in place; also emit bf16 copy for next GEMM.
__global__ void k_bias_act(float* __restrict__ feat, const float* __restrict__ b,
                           unsigned short* __restrict__ featB, int total) {
  int i = blockIdx.x * blockDim.x + threadIdx.x;
  if (i >= total) return;
  float v = feat[i] + b[i & 255];
  v = v > 0.f ? v : 0.01f * v;
  feat[i]  = v;
  featB[i] = f2bf(v);
}

// out[n] = <feat[n,:], cls_W> + cls_b. One wave per node.
__global__ void k_classifier(const float* __restrict__ feat, const float* __restrict__ w,
                             const float* __restrict__ b, float* __restrict__ out, int N) {
  int n    = (blockIdx.x * blockDim.x + threadIdx.x) >> 5;
  int lane = threadIdx.x & 31;
  if (n >= N) return;
  const float* f = feat + (size_t)n * 256;
  float s = 0.f;
#pragma unroll
  for (int i = 0; i < 8; ++i) s += f[lane + i * 32] * w[lane + i * 32];
  for (int o = 16; o > 0; o >>= 1) s += __shfl_down(s, o);
  if (lane == 0) out[n] = s + b[0];
}

// ---------------------------------------------------------------------------
extern "C" void kernel_launch(void* const* d_in, const int* in_sizes, int n_in,
                              void* d_out, int out_size, void* d_ws, size_t ws_size,
                              hipStream_t stream) {
  const float* x     = (const float*)d_in[0];
  const int*   ei    = (const int*)d_in[1];      // [2,E]: row0=src, row1=dst
  const int*   batch = (const int*)d_in[2];
  const int*   clab  = (const int*)d_in[3];
  const float* emb   = (const float*)d_in[4];
  const float* Wl[3]   = {(const float*)d_in[5],  (const float*)d_in[9],  (const float*)d_in[13]};
  const float* asrc[3] = {(const float*)d_in[6],  (const float*)d_in[10], (const float*)d_in[14]};
  const float* adst[3] = {(const float*)d_in[7],  (const float*)d_in[11], (const float*)d_in[15]};
  const float* bl[3]   = {(const float*)d_in[8],  (const float*)d_in[12], (const float*)d_in[16]};
  const float* clsW = (const float*)d_in[17];
  const float* clsb = (const float*)d_in[18];

  const int N    = in_sizes[0] / 64;
  const int E    = in_sizes[1] / 2;
  const int Etot = E + N;

  // ---- workspace carve-up (~134 MB) ----
  char*  ws  = (char*)d_ws;
  size_t off = 0;
  auto alloc = [&](size_t bytes) -> void* {
    off = (off + 255) & ~(size_t)255;
    void* p = ws + off;
    off += bytes;
    return p;
  };
  unsigned short* featB = (unsigned short*)alloc((size_t)N * 256 * 2); // bf16 features
  unsigned short* wB    = (unsigned short*)alloc((size_t)256 * 256 * 2);
  float* hbuf  = (float*)alloc((size_t)N * 256 * 4);  // projected h (per layer)
  float* feat  = (float*)alloc((size_t)N * 256 * 4);  // aggregated out / next feats
  float* ssrc  = (float*)alloc((size_t)N * 4 * 4);
  float* sdst  = (float*)alloc((size_t)N * 4 * 4);
  float* mmax  = (float*)alloc((size_t)N * 4 * 4);
  float* denom = (float*)alloc((size_t)N * 4 * 4);
  (void)ws_size; (void)n_in; (void)out_size;

  const int TB = 256;
  auto nb = [](long long t, int b) { return (int)((t + b - 1) / b); };
  const float NEG_INF = -__builtin_huge_valf();

  k_concat<<<nb((long long)N * 128, TB), TB, 0, stream>>>(x, batch, clab, emb, featB, N);

  const int Kdim[3] = {128, 256, 256};
  for (int l = 0; l < 3; ++l) {
    const int K = Kdim[l];
    k_cvt_bf16<<<nb(256 * K, TB), TB, 0, stream>>>(Wl[l], wB, 256 * K);

    const int mtiles = (N + 15) / 16;
    const long long waves = (long long)mtiles * (256 / 16);
    k_gemm_bf16<<<nb(waves * 32, TB), TB, 0, stream>>>(featB, wB, hbuf, N, K, 256);

    k_scores<<<nb((long long)N * 4 * 32, TB), TB, 0, stream>>>(
        hbuf, asrc[l], adst[l], ssrc, sdst, N);

    k_fill<<<nb(N * 4, TB), TB, 0, stream>>>(mmax, NEG_INF, N * 4);
    k_fill<<<nb(N * 4, TB), TB, 0, stream>>>(denom, 0.f, N * 4);
    k_fill<<<nb((long long)N * 256, TB), TB, 0, stream>>>(feat, 0.f, N * 256);

    k_alpha_max<<<nb((long long)Etot * 4, TB), TB, 0, stream>>>(ei, ssrc, sdst, mmax, E, Etot * 4);
    k_exp_sum <<<nb((long long)Etot * 4, TB), TB, 0, stream>>>(ei, ssrc, sdst, mmax, denom, E, Etot * 4);
    k_aggregate<<<Etot, TB, 0, stream>>>(ei, ssrc, sdst, mmax, denom, hbuf, feat, E);

    k_bias_act<<<nb((long long)N * 256, TB), TB, 0, stream>>>(feat, bl[l], featB, N * 256);
  }

  k_classifier<<<nb((long long)N * 32, TB), TB, 0, stream>>>(feat, clsW, clsb, (float*)d_out, N);
}
